// SpliceGraph_88149908783398
// MI455X (gfx1250) — matrix operand order repaired and checked by reading, not compiled, not run on hardware
//
#include <hip/hip_runtime.h>
#include <hip/hip_bf16.h>
#include <math.h>

// ---------------------------------------------------------------------------
// Types for CDNA5 WMMA (wave32)
// ---------------------------------------------------------------------------
typedef __attribute__((ext_vector_type(16))) __bf16     v16bf;
typedef __attribute__((ext_vector_type(8)))  float      v8f;
typedef __attribute__((ext_vector_type(8)))  unsigned   v8u;

#define D_HID 256
#define BN_EPS 1e-5f

// float -> bf16 (round to nearest even), packed pair
__device__ __forceinline__ unsigned bf16r(float f) {
    unsigned u = __float_as_uint(f);
    return (u + 0x7FFFu + ((u >> 16) & 1u)) >> 16;
}
__device__ __forceinline__ unsigned pack_bf2(float lo, float hi) {
    return bf16r(lo) | (bf16r(hi) << 16);
}

__device__ __forceinline__ void atomAddF(float* p, float v) {
    __hip_atomic_fetch_add(p, v, __ATOMIC_RELAXED, __HIP_MEMORY_SCOPE_AGENT);
}

// ---------------------------------------------------------------------------
// Pre-swizzle a 256x256 f32 weight matrix into the CDNA5 WMMA B-fragment
// layout as packed bf16 pairs (u32). Index layout:
//   out[((ks*16 + ntile)*32 + lane)*8 + v]
//   lane<16  : n = ntile*16 + lane,    K = ks*32 + 2(v&3) + 16*(v>>2) (+1 in high half)
//   lane>=16 : n = ntile*16 + lane-16, K = ks*32 + 8 + 2(v&3) + 16*(v>>2) (+1 high)
// so each wave reads a B fragment as 8 contiguous u32 per lane (2x b128).
// ---------------------------------------------------------------------------
__global__ void k_prepW(const float* __restrict__ W, unsigned* __restrict__ out) {
    int idx  = blockIdx.x * 256 + threadIdx.x;    // 0..32767
    int v    = idx & 7;
    int lane = (idx >> 3) & 31;
    int nt   = (idx >> 8) & 15;
    int ks   = idx >> 12;
    int K    = ks * 32 + ((lane >> 4) << 3) + 2 * (v & 3) + ((v >> 2) << 4);
    int n    = nt * 16 + (lane & 15);
    out[idx] = pack_bf2(W[(size_t)K * D_HID + n], W[(size_t)(K + 1) * D_HID + n]);
}

// ---------------------------------------------------------------------------
// Fused WMMA GEMM: C[N,256] = act(A[N,256] @ W + ...)
// Block = 256 threads = 8 waves; block computes 32 rows x 256 cols.
// Wave w: m-tile = w&1, n-tiles = (w>>1)*4 .. +3  -> 4 accumulators.
// A tile staged in LDS as packed bf16 pairs (32 rows x 128 u32 = 16KB).
// MODE 0: C = acc
// MODE 1: C = sigmoid(acc + bias)
// MODE 2: xl = acc + bias; C = relu((1-g)*xl + g*z)
// ---------------------------------------------------------------------------
template <int MODE>
__global__ __launch_bounds__(256) void k_gemm(
    const float*    __restrict__ A,
    const unsigned* __restrict__ Wf,
    const float*    __restrict__ bias,
    const float*    __restrict__ Zb,
    const float*    __restrict__ Gb,
    float*          __restrict__ Cout)
{
    __shared__ unsigned lds[32 * 128];
    const int tid   = threadIdx.x;
    const int mrow0 = blockIdx.x * 32;

    // ---- stage A tile (f32 -> packed bf16) ----
    const float2* A2 = reinterpret_cast<const float2*>(A);
#pragma unroll
    for (int i = 0; i < 16; ++i) {
        int idx = tid + 256 * i;
        int r   = idx >> 7;
        int c   = idx & 127;
        float2 f = A2[(size_t)(mrow0 + r) * 128 + c];
        lds[idx] = pack_bf2(f.x, f.y);
    }
    __syncthreads();

    const int lane = tid & 31;
    const int wave = tid >> 5;
    const int mt   = wave & 1;          // which 16-row half
    const int nt0  = (wave >> 1) << 2;  // first of 4 n-tiles

    v8f acc[4];
    {
        v8f z8 = {0.f, 0.f, 0.f, 0.f, 0.f, 0.f, 0.f, 0.f};
#pragma unroll
        for (int i = 0; i < 4; ++i) acc[i] = z8;
    }

    const v8u* Wv = reinterpret_cast<const v8u*>(Wf);
    const int  rl = mt * 16 + (lane & 15);     // A row within LDS tile

#pragma unroll
    for (int ks = 0; ks < 8; ++ks) {
        // A fragment (16x32 bf16): lane group selects K half per ISA layout
        int cb = ks * 16 + ((lane >> 4) << 2); // u32 column base (K/2)
        v8u au;
        au[0] = lds[rl * 128 + cb + 0];
        au[1] = lds[rl * 128 + cb + 1];
        au[2] = lds[rl * 128 + cb + 2];
        au[3] = lds[rl * 128 + cb + 3];
        au[4] = lds[rl * 128 + cb + 8];
        au[5] = lds[rl * 128 + cb + 9];
        au[6] = lds[rl * 128 + cb + 10];
        au[7] = lds[rl * 128 + cb + 11];
        v16bf a = __builtin_bit_cast(v16bf, au);
#pragma unroll
        for (int i = 0; i < 4; ++i) {
            int  nt = nt0 + i;
            v8u  bu = Wv[(size_t)(ks * 16 + nt) * 32 + lane];
            v16bf b = __builtin_bit_cast(v16bf, bu);
            acc[i] = __builtin_amdgcn_wmma_f32_16x16x32_bf16(
                false, a, false, b, (short)0, acc[i], false, false);
        }
    }

    // ---- epilogue: C/D layout: lane<16 -> M=v, lane>=16 -> M=v+8; N=lane%16
    const int nIdx = lane & 15;
    const int mh   = (lane >> 4) * 8;
#pragma unroll
    for (int i = 0; i < 4; ++i) {
        int colg = (nt0 + i) * 16 + nIdx;
#pragma unroll
        for (int v = 0; v < 8; ++v) {
            int    row = mrow0 + mt * 16 + mh + v;
            size_t o   = (size_t)row * D_HID + colg;
            float  val = acc[i][v];
            if (MODE == 0) {
                Cout[o] = val;
            } else if (MODE == 1) {
                float t = val + bias[colg];
                Cout[o] = 1.f / (1.f + expf(-t));
            } else {
                float xl = val + bias[colg];
                float g  = Gb[o];
                float zz = Zb[o];
                float h  = (1.f - g) * xl + g * zz;
                Cout[o]  = h > 0.f ? h : 0.f;
            }
        }
    }
}

// ---------------------------------------------------------------------------
// Degree / normalization helpers
// ---------------------------------------------------------------------------
__global__ void k_initdeg(unsigned* deg, int N) {
    int i = blockIdx.x * blockDim.x + threadIdx.x;
    if (i < N) deg[i] = 1u;                  // self-loop
}
__global__ void k_deg(const int* __restrict__ col, unsigned* deg, int E) {
    int e = blockIdx.x * blockDim.x + threadIdx.x;
    if (e < E) atomicAdd(&deg[col[e]], 1u);
}
__global__ void k_dinv(const unsigned* __restrict__ deg, float* __restrict__ dinv, int N) {
    int i = blockIdx.x * blockDim.x + threadIdx.x;
    if (i < N) dinv[i] = rsqrtf((float)deg[i]);
}
__global__ void k_zerof(float* p, long n) {
    long i = (long)blockIdx.x * blockDim.x + threadIdx.x;
    if (i < n) p[i] = 0.f;
}

// ---------------------------------------------------------------------------
// Edge scatter: one wave (32 lanes) per edge; lane handles 8 columns via
// two float4 loads; fp32 atomic adds resolve in L2 (z fits in 192MB L2).
// ---------------------------------------------------------------------------
__global__ __launch_bounds__(256) void k_scatter(
    const int* __restrict__ ei, const float* __restrict__ xw,
    const float* __restrict__ dinv, float* __restrict__ z, int E)
{
    int gtid = blockIdx.x * blockDim.x + threadIdx.x;
    int e    = gtid >> 5;
    int lane = threadIdx.x & 31;
    if (e >= E) return;
    int   r   = ei[e];
    int   c   = ei[E + e];
    float nrm = dinv[r] * dinv[c];
    const float4* src = reinterpret_cast<const float4*>(xw + (size_t)r * D_HID);
    float*        dst = z + (size_t)c * D_HID;
    float4 a = src[lane * 2 + 0];
    float4 b = src[lane * 2 + 1];
    int base = lane * 8;
    atomAddF(dst + base + 0, a.x * nrm);
    atomAddF(dst + base + 1, a.y * nrm);
    atomAddF(dst + base + 2, a.z * nrm);
    atomAddF(dst + base + 3, a.w * nrm);
    atomAddF(dst + base + 4, b.x * nrm);
    atomAddF(dst + base + 5, b.y * nrm);
    atomAddF(dst + base + 6, b.z * nrm);
    atomAddF(dst + base + 7, b.w * nrm);
}

// z = tanh(z_accum + dinv^2 * xw  (self-loop)  + gcn_b)
__global__ void k_tanh(float* __restrict__ z, const float* __restrict__ xw,
                       const float* __restrict__ dinv, const float* __restrict__ b,
                       long total)
{
    long i = (long)blockIdx.x * blockDim.x + threadIdx.x;
    if (i >= total) return;
    int node = (int)(i >> 8);
    int c    = (int)(i & 255);
    float d  = dinv[node];
    z[i] = tanhf(z[i] + d * d * xw[i] + b[c]);
}

// ---------------------------------------------------------------------------
// BatchNorm: column-per-thread block reduction (coalesced), then apply.
// ---------------------------------------------------------------------------
__global__ __launch_bounds__(256) void k_bnstats(
    const float* __restrict__ h, float* __restrict__ sum, float* __restrict__ sumsq, int N)
{
    int c  = threadIdx.x;
    int r0 = blockIdx.x * 256;
    int r1 = r0 + 256 < N ? r0 + 256 : N;
    float s = 0.f, s2 = 0.f;
    for (int r = r0; r < r1; ++r) {
        float v = h[(size_t)r * D_HID + c];
        s += v;
        s2 += v * v;
    }
    atomAddF(&sum[c], s);
    atomAddF(&sumsq[c], s2);
}

__global__ void k_bnapply(float* __restrict__ h, const float* __restrict__ sum,
                          const float* __restrict__ sumsq,
                          const float* __restrict__ gamma, const float* __restrict__ beta,
                          long total, float invN)
{
    long i = (long)blockIdx.x * blockDim.x + threadIdx.x;
    if (i >= total) return;
    int   c    = (int)(i & 255);
    float mean = sum[c] * invN;
    float var  = sumsq[c] * invN - mean * mean;
    h[i] = (h[i] - mean) * rsqrtf(var + BN_EPS) * gamma[c] + beta[c];
}

// ---------------------------------------------------------------------------
// Launcher
// ---------------------------------------------------------------------------
extern "C" void kernel_launch(void* const* d_in, const int* in_sizes, int n_in,
                              void* d_out, int out_size, void* d_ws, size_t ws_size,
                              hipStream_t stream) {
    (void)n_in; (void)out_size; (void)ws_size;

    const int N = in_sizes[0] / D_HID;   // 100000
    const int E = in_sizes[1] / 2;       // 1600000

    const float* x      = (const float*)d_in[0];
    const int*   ei     = (const int*)d_in[1];   // JAX x64 disabled -> int32
    const float* gcn_W  = (const float*)d_in[2];
    const float* gcn_b  = (const float*)d_in[3];
    const float* lin_W  = (const float*)d_in[4];
    const float* lin_b  = (const float*)d_in[5];
    const float* gate_W = (const float*)d_in[6];
    const float* gate_b = (const float*)d_in[7];
    const float* gamma  = (const float*)d_in[8];
    const float* beta   = (const float*)d_in[9];
    float*       out    = (float*)d_out;

    // workspace layout
    float*    xw      = (float*)d_ws;                       // N*256 (later reused for g)
    float*    z       = xw + (size_t)N * D_HID;             // N*256
    unsigned* wf_gcn  = (unsigned*)(z + (size_t)N * D_HID); // 32768 u32 each
    unsigned* wf_gate = wf_gcn + 32768;
    unsigned* wf_lin  = wf_gate + 32768;
    unsigned* deg     = wf_lin + 32768;                     // N u32
    float*    dinv    = (float*)(deg + N);                  // N f32
    float*    bnsum   = dinv + N;                           // 256 f32
    float*    bnsq    = bnsum + 256;                        // 256 f32 (contiguous)

    const long total = (long)N * D_HID;
    const int  TPB   = 256;

    // weight pre-swizzle to WMMA B-fragment bf16 layout
    k_prepW<<<128, TPB, 0, stream>>>(gcn_W, wf_gcn);
    k_prepW<<<128, TPB, 0, stream>>>(gate_W, wf_gate);
    k_prepW<<<128, TPB, 0, stream>>>(lin_W, wf_lin);

    // degrees (with self-loop) and D^-1/2
    k_initdeg<<<(N + TPB - 1) / TPB, TPB, 0, stream>>>(deg, N);
    k_deg<<<(E + TPB - 1) / TPB, TPB, 0, stream>>>(ei + E, deg, E);
    k_dinv<<<(N + TPB - 1) / TPB, TPB, 0, stream>>>(deg, dinv, N);

    // zero accumulators (ws is poisoned by harness)
    k_zerof<<<(int)((total + TPB - 1) / TPB), TPB, 0, stream>>>(z, total);
    k_zerof<<<2, TPB, 0, stream>>>(bnsum, 512);

    // xw = x @ gcn_W   (bf16 WMMA, f32 accumulate)
    k_gemm<0><<<N / 32, TPB, 0, stream>>>(x, wf_gcn, nullptr, nullptr, nullptr, xw);

    // z += norm-scaled neighbor messages (L2-resident gather + fp32 atomics)
    k_scatter<<<(E + 7) / 8, TPB, 0, stream>>>(ei, xw, dinv, z, E);

    // z = tanh(z + self-loop + bias)
    k_tanh<<<(int)((total + TPB - 1) / TPB), TPB, 0, stream>>>(z, xw, dinv, gcn_b, total);

    // g = sigmoid(z @ gate_W + gate_b)   (reuse xw buffer for g)
    k_gemm<1><<<N / 32, TPB, 0, stream>>>(z, wf_gate, gate_b, nullptr, nullptr, xw);

    // h = relu((1-g)*(x @ lin_W + lin_b) + g*z)  -> d_out
    k_gemm<2><<<N / 32, TPB, 0, stream>>>(x, wf_lin, lin_b, z, xw, out);

    // BatchNorm (training-mode batch stats)
    k_bnstats<<<(N + 255) / 256, 256, 0, stream>>>(out, bnsum, bnsq, N);
    k_bnapply<<<(int)((total + TPB - 1) / TPB), TPB, 0, stream>>>(
        out, bnsum, bnsq, gamma, beta, total, 1.0f / (float)N);
}
